// SlotAttention_13657996001922
// MI455X (gfx1250) — matrix operand order repaired and checked by reading, not compiled, not run on hardware
//
#include <hip/hip_runtime.h>
#include <hip/hip_bf16.h>

typedef __attribute__((ext_vector_type(16))) _Float16 v16h;
typedef __attribute__((ext_vector_type(8)))  _Float16 v8h;
typedef __attribute__((ext_vector_type(2)))  _Float16 v2h;
typedef __attribute__((ext_vector_type(8)))  float    v8f;
typedef int v4i __attribute__((vector_size(16)));

#define BN 64
#define CN 128
#define NP 4096
#define DD 64
#define SS 7
#define ITERS 3
#define EPS 1e-5f

#define XH_P 136   // padded row stride (halfs) for f16 activation/weight tiles
#define VR_P 72    // padded row stride (halfs) for v tile (16B-aligned rows)
#define QH_P 72
#define AH_P 40

static __device__ __forceinline__ v8f wmma_f16(v16h a, v16h b, v8f c) {
    return __builtin_amdgcn_wmma_f32_16x16x32_f16(false, a, false, b, (short)0, c, false, false);
}

// 16-byte global -> LDS copy: async (ASYNCcnt) on gfx1250 toolchains that
// expose the builtin, else a synchronous vector copy.
static __device__ __forceinline__ void copy16_g2l(const _Float16* g, _Float16* l) {
#if __has_builtin(__builtin_amdgcn_global_load_async_to_lds_b128)
    __builtin_amdgcn_global_load_async_to_lds_b128(
        (v4i __attribute__((address_space(1)))*)(g),
        (v4i __attribute__((address_space(3)))*)(l), 0, 0);
#else
    *(v8h*)l = *(const v8h*)g;
#endif
}

static __device__ __forceinline__ void wait_async_zero() {
#if __has_builtin(__builtin_amdgcn_s_wait_asynccnt)
    __builtin_amdgcn_s_wait_asynccnt(0);
#else
    asm volatile("s_wait_asynccnt 0" ::: "memory");
#endif
}

// ---------------------------------------------------------------------------
// Kernel 1: transpose + LayerNorm(C) + K/V projection  (WMMA f16 -> f32)
// grid (N/128, B), block 256 (8 waves). Each wave: 16 pixels x 64 outputs x2.
// ---------------------------------------------------------------------------
__global__ __launch_bounds__(256) void kv_proj_kernel(
    const float* __restrict__ x,   const float* __restrict__ lng,
    const float* __restrict__ lnb, const float* __restrict__ Wk,
    const float* __restrict__ Wv,
    _Float16* __restrict__ kbuf,   _Float16* __restrict__ vbuf)
{
    __shared__ float    xs[128][129];       // raw [pixel][chan] (padded, f32)
    __shared__ _Float16 xh[128][XH_P];      // normalized f16, padded
    __shared__ _Float16 wk[DD][XH_P];
    __shared__ _Float16 wv[DD][XH_P];

    const int b     = blockIdx.y;
    const int nbase = blockIdx.x * 128;
    const int tid   = threadIdx.x;

    for (int e = tid; e < DD * CN; e += 256) {
        int d = e >> 7, c = e & 127;
        wk[d][c] = (_Float16)Wk[e];
        wv[d][c] = (_Float16)Wv[e];
    }
    // coalesced float4 transpose load: x is [B, C, N]
    const float* xb = x + (size_t)b * CN * NP + nbase;
    for (int e = tid; e < 128 * 32; e += 256) {
        int c  = e >> 5;
        int p4 = (e & 31) << 2;
        float4 vld = *(const float4*)(xb + (size_t)c * NP + p4);
        xs[p4 + 0][c] = vld.x;
        xs[p4 + 1][c] = vld.y;
        xs[p4 + 2][c] = vld.z;
        xs[p4 + 3][c] = vld.w;
    }
    __syncthreads();

    // per-pixel LayerNorm over C=128; emit normalized row as f16 pairs
    if (tid < 128) {
        float m = 0.f;
        for (int c = 0; c < CN; ++c) m += xs[tid][c];
        m *= (1.0f / CN);
        float v = 0.f;
        for (int c = 0; c < CN; ++c) { float d = xs[tid][c] - m; v += d * d; }
        v *= (1.0f / CN);
        float r = rsqrtf(v + EPS);
        for (int c = 0; c < CN; c += 2) {
            float n0 = (xs[tid][c]     - m) * r * lng[c]     + lnb[c];
            float n1 = (xs[tid][c + 1] - m) * r * lng[c + 1] + lnb[c + 1];
            v2h pk; pk[0] = (_Float16)n0; pk[1] = (_Float16)n1;
            *(v2h*)&xh[tid][c] = pk;
        }
    }
    __syncthreads();

    const int w    = tid >> 5;
    const int ln   = tid & 31;
    const int m0   = w * 16;
    const int lrow = m0 + (ln & 15);
    const int hi   = ln >> 4;

    v8f acck[4], accv[4];
#pragma unroll
    for (int dt = 0; dt < 4; ++dt)
#pragma unroll
        for (int i = 0; i < 8; ++i) { acck[dt][i] = 0.f; accv[dt][i] = 0.f; }

#pragma unroll
    for (int ks = 0; ks < 4; ++ks) {
        const int kk0 = ks * 32;
        // A fragment 16x32 f16: 2 x ds_load_b128 of packed halfs
        const int kbA = kk0 + (hi << 4);
        v8h alo = *(const v8h*)&xh[lrow][kbA];
        v8h ahi = *(const v8h*)&xh[lrow][kbA + 8];
        v16h a;
#pragma unroll
        for (int j = 0; j < 8; ++j) { a[j] = alo[j]; a[j + 8] = ahi[j]; }

        const int col = ln & 15;
        const int kbB = kk0 + (hi << 4);
#pragma unroll
        for (int dt = 0; dt < 4; ++dt) {
            const int d = dt * 16 + col;
            v8h bk0 = *(const v8h*)&wk[d][kbB];
            v8h bk1 = *(const v8h*)&wk[d][kbB + 8];
            v8h bv0 = *(const v8h*)&wv[d][kbB];
            v8h bv1 = *(const v8h*)&wv[d][kbB + 8];
            v16h bk, bv;
#pragma unroll
            for (int j = 0; j < 8; ++j) {
                bk[j] = bk0[j]; bk[j + 8] = bk1[j];
                bv[j] = bv0[j]; bv[j + 8] = bv1[j];
            }
            acck[dt] = wmma_f16(a, bk, acck[dt]);
            accv[dt] = wmma_f16(a, bv, accv[dt]);
        }
    }

    // store K/V tiles as f16, layout [B, N, D]
    const size_t base = ((size_t)b * NP + nbase) * DD;
#pragma unroll
    for (int dt = 0; dt < 4; ++dt) {
        const int d = dt * 16 + (ln & 15);
#pragma unroll
        for (int i = 0; i < 8; ++i) {
            const int m = m0 + i + (hi << 3);
            kbuf[base + (size_t)m * DD + d] = (_Float16)acck[dt][i];
            vbuf[base + (size_t)m * DD + d] = (_Float16)accv[dt][i];
        }
    }
}

// ---------------------------------------------------------------------------
// Kernel 2: 3 slot-attention iterations fused; one workgroup per batch image.
// WMMA for logits (k @ q^T, S padded to 16) and updates (attn^T @ v).
// v chunks double-buffered in LDS via async global->LDS copies (ASYNCcnt).
// ---------------------------------------------------------------------------
__global__ __launch_bounds__(256) void slot_iter_kernel(
    const _Float16* __restrict__ kbuf, const _Float16* __restrict__ vbuf,
    const float* __restrict__ slots_mu,
    const float* __restrict__ ln_s_g, const float* __restrict__ ln_s_b,
    const float* __restrict__ Wq,
    const float* __restrict__ W_ih, const float* __restrict__ W_hh,
    const float* __restrict__ b_ih, const float* __restrict__ b_hh,
    const float* __restrict__ ln_m_g, const float* __restrict__ ln_m_b,
    const float* __restrict__ W1, const float* __restrict__ b1,
    const float* __restrict__ W2, const float* __restrict__ b2,
    float* __restrict__ out)
{
    __shared__ float    slots[SS][DD];
    __shared__ float    sn[SS][DD];          // LN(slots) / LN(h)
    __shared__ _Float16 qh[16][QH_P];        // q^T operand, pre-scaled, rows 7..15 zero
    __shared__ float    lbuf[256][17];       // logits per pass (padded)
    __shared__ _Float16 ah[8][16][AH_P];     // attn^T per wave (rows 7..15 zero)
    __shared__ _Float16 vr[2][256][VR_P];    // double-buffered v chunk [pixel][d]
    __shared__ float    ubuf[8][SS][DD];     // per-wave partial updates
    __shared__ float    uf[SS][DD];          // reduced updates
    __shared__ float    gi[SS][192];
    __shared__ float    gh[SS][192];
    __shared__ float    hbuf[SS][DD];
    __shared__ float    t1[SS][DD];

    const int b   = blockIdx.x;
    const int tid = threadIdx.x;
    const int w   = tid >> 5;
    const int ln  = tid & 31;
    const int hi  = ln >> 4;

    for (int e = tid; e < SS * DD; e += 256)
        (&slots[0][0])[e] = slots_mu[e];
    __syncthreads();

    const _Float16* kb_b = kbuf + (size_t)b * NP * DD;
    const _Float16* vb_b = vbuf + (size_t)b * NP * DD;

    for (int it = 0; it < ITERS; ++it) {
        // LN(slots) over D=64 -> sn
        if (tid < SS) {
            float m = 0.f;
            for (int d = 0; d < DD; ++d) m += slots[tid][d];
            m *= (1.0f / DD);
            float v = 0.f;
            for (int d = 0; d < DD; ++d) { float t = slots[tid][d] - m; v += t * t; }
            v *= (1.0f / DD);
            float r = rsqrtf(v + EPS);
            for (int d = 0; d < DD; ++d)
                sn[tid][d] = (slots[tid][d] - m) * r * ln_s_g[d] + ln_s_b[d];
        }
        __syncthreads();
        // q = LN(slots) @ Wq^T, fold in scale = D^-0.5 = 0.125; f16, padded rows
        for (int e = tid; e < 16 * DD; e += 256) {
            int s = e >> 6, dd = e & 63;
            float acc = 0.f;
            if (s < SS) {
                for (int d = 0; d < DD; ++d) acc += sn[s][d] * Wq[dd * DD + d];
                acc *= 0.125f;
            }
            qh[s][dd] = (_Float16)acc;
        }
        __syncthreads();

        v8f upd[4];
#pragma unroll
        for (int dt = 0; dt < 4; ++dt)
#pragma unroll
            for (int i = 0; i < 8; ++i) upd[dt][i] = 0.f;

        // prologue: stage pass 0's v chunk into buffer 0 (async)
#pragma unroll
        for (int i = 0; i < 8; ++i) {
            int flat8 = tid + i * 256;            // 2048 x 8-half segments
            int p  = flat8 >> 3;
            int c0 = (flat8 & 7) << 3;
            copy16_g2l(vb_b + (size_t)p * DD + c0, &vr[0][p][c0]);
        }

        for (int pass = 0; pass < 16; ++pass) {
            const int pbase = pass * 256;
            const int pw    = pbase + w * 32;     // this wave's 32 pixels
            const int cur   = pass & 1;
            const int nxt   = cur ^ 1;

            wait_async_zero();      // this wave's copies into vr[cur] arrived
            __syncthreads();        // all waves' copies arrived; prior readers of
                                    // vr[nxt] / lbuf / ah are done

            // stage next pass's v chunk while this pass computes
            if (pass < 15) {
#pragma unroll
                for (int i = 0; i < 8; ++i) {
                    int flat8 = tid + i * 256;
                    int p  = flat8 >> 3;
                    int c0 = (flat8 & 7) << 3;
                    copy16_g2l(vb_b + (size_t)(pbase + 256 + p) * DD + c0,
                               &vr[nxt][p][c0]);
                }
                __builtin_prefetch(kb_b + (size_t)(pw + 256) * DD + ln * 64, 0, 3);
            }

            // ---- logits = k @ q^T (scaled): two 16-row tiles per wave ----
#pragma unroll
            for (int mt = 0; mt < 2; ++mt) {
                v8f lacc;
#pragma unroll
                for (int i = 0; i < 8; ++i) lacc[i] = 0.f;
#pragma unroll
                for (int ks = 0; ks < 2; ++ks) {
                    const int row = pw + mt * 16 + (ln & 15);
                    const int kbA = ks * 32 + (hi << 4);
                    const _Float16* kp = kb_b + (size_t)row * DD + kbA;
                    v8h klo = *(const v8h*)kp;
                    v8h khi = *(const v8h*)(kp + 8);
                    v16h a;
#pragma unroll
                    for (int j = 0; j < 8; ++j) { a[j] = klo[j]; a[j + 8] = khi[j]; }

                    const int col = ln & 15;
                    const int kbB = ks * 32 + (hi << 4);
                    v8h q0 = *(const v8h*)&qh[col][kbB];
                    v8h q1 = *(const v8h*)&qh[col][kbB + 8];
                    v16h bq;
#pragma unroll
                    for (int j = 0; j < 8; ++j) { bq[j] = q0[j]; bq[j + 8] = q1[j]; }
                    lacc = wmma_f16(a, bq, lacc);
                }
#pragma unroll
                for (int i = 0; i < 8; ++i) {
                    const int m = mt * 16 + i + (hi << 3);
                    lbuf[w * 32 + m][ln & 15] = lacc[i];
                }
            }
            __syncthreads();

            // ---- softmax over S=7 per pixel; write attn^T in A-fragment layout ----
            {
                float lg[SS];
                float mx = -3.4e38f;
#pragma unroll
                for (int s = 0; s < SS; ++s) { lg[s] = lbuf[tid][s]; mx = fmaxf(mx, lg[s]); }
                float sum = 0.f;
#pragma unroll
                for (int s = 0; s < SS; ++s) { lg[s] = __expf(lg[s] - mx); sum += lg[s]; }
                const float inv = 1.0f / sum;
                const int nl = tid & 31;
#pragma unroll
                for (int s = 0; s < 16; ++s)
                    ah[tid >> 5][s][nl] = (s < SS) ? (_Float16)(lg[s] * inv) : (_Float16)0.f;
            }
            __syncthreads();

            // ---- updates += attn^T @ v  (M = S padded 16, K = 32 pixels) ----
            {
                const int srow = ln & 15;
                const int kbA  = hi << 4;
                v8h a0 = *(const v8h*)&ah[w][srow][kbA];
                v8h a1 = *(const v8h*)&ah[w][srow][kbA + 8];
                v16h a;
#pragma unroll
                for (int j = 0; j < 8; ++j) { a[j] = a0[j]; a[j + 8] = a1[j]; }

                const int col    = ln & 15;
                const int kbB    = hi << 4;
                const int plocal = w * 32;
#pragma unroll
                for (int dt = 0; dt < 4; ++dt) {
                    v16h bv;
#pragma unroll
                    for (int j = 0; j < 16; ++j)
                        bv[j] = vr[cur][plocal + kbB + j][dt * 16 + col];
                    upd[dt] = wmma_f16(a, bv, upd[dt]);
                }
            }
            // no end-of-pass barrier: next pass's start barrier orders reuse
        }

        // ---- reduce per-wave partial updates ----
        __syncthreads();
#pragma unroll
        for (int dt = 0; dt < 4; ++dt)
#pragma unroll
            for (int i = 0; i < 8; ++i) {
                const int s = i + (hi << 3);
                if (s < SS) ubuf[w][s][dt * 16 + (ln & 15)] = upd[dt][i];
            }
        __syncthreads();
        for (int e = tid; e < SS * DD; e += 256) {
            int s = e >> 6, d = e & 63;
            float acc = 0.f;
#pragma unroll
            for (int ww = 0; ww < 8; ++ww) acc += ubuf[ww][s][d];
            uf[s][d] = acc;
        }
        __syncthreads();

        // ---- GRU gates: gi = updates @ W_ih^T + b_ih ; gh = slots @ W_hh^T + b_hh
        for (int e = tid; e < SS * 192; e += 256) {
            int s = e / 192, c = e % 192;
            float a1 = b_ih[c], a2 = b_hh[c];
            for (int d = 0; d < DD; ++d) {
                a1 += uf[s][d]    * W_ih[c * DD + d];
                a2 += slots[s][d] * W_hh[c * DD + d];
            }
            gi[s][c] = a1; gh[s][c] = a2;
        }
        __syncthreads();
        for (int e = tid; e < SS * DD; e += 256) {
            int s = e >> 6, d = e & 63;
            float r  = 1.0f / (1.0f + __expf(-(gi[s][d]       + gh[s][d])));
            float z  = 1.0f / (1.0f + __expf(-(gi[s][64 + d]  + gh[s][64 + d])));
            float nn = tanhf(gi[s][128 + d] + r * gh[s][128 + d]);
            hbuf[s][d] = (1.0f - z) * nn + z * slots[s][d];
        }
        __syncthreads();

        // ---- LN(h) -> sn ----
        if (tid < SS) {
            float m = 0.f;
            for (int d = 0; d < DD; ++d) m += hbuf[tid][d];
            m *= (1.0f / DD);
            float v = 0.f;
            for (int d = 0; d < DD; ++d) { float t = hbuf[tid][d] - m; v += t * t; }
            v *= (1.0f / DD);
            float r = rsqrtf(v + EPS);
            for (int d = 0; d < DD; ++d)
                sn[tid][d] = (hbuf[tid][d] - m) * r * ln_m_g[d] + ln_m_b[d];
        }
        __syncthreads();

        // ---- MLP: slots = h + relu(sn @ W1^T + b1) @ W2^T + b2 ----
        for (int e = tid; e < SS * DD; e += 256) {
            int s = e >> 6, j = e & 63;
            float acc = b1[j];
            for (int d = 0; d < DD; ++d) acc += sn[s][d] * W1[j * DD + d];
            t1[s][j] = fmaxf(acc, 0.f);
        }
        __syncthreads();
        for (int e = tid; e < SS * DD; e += 256) {
            int s = e >> 6, d = e & 63;
            float acc = b2[d];
            for (int j = 0; j < DD; ++j) acc += t1[s][j] * W2[d * DD + j];
            slots[s][d] = hbuf[s][d] + acc;
        }
        __syncthreads();
    }

    for (int e = tid; e < SS * DD; e += 256)
        out[(size_t)b * SS * DD + e] = (&slots[0][0])[e];
}

// ---------------------------------------------------------------------------
extern "C" void kernel_launch(void* const* d_in, const int* in_sizes, int n_in,
                              void* d_out, int out_size, void* d_ws, size_t ws_size,
                              hipStream_t stream) {
    const float* x        = (const float*)d_in[0];
    const float* slots_mu = (const float*)d_in[1];
    const float* ln_in_g  = (const float*)d_in[2];
    const float* ln_in_b  = (const float*)d_in[3];
    const float* Wk       = (const float*)d_in[4];
    const float* Wv       = (const float*)d_in[5];
    const float* ln_s_g   = (const float*)d_in[6];
    const float* ln_s_b   = (const float*)d_in[7];
    const float* Wq       = (const float*)d_in[8];
    const float* W_ih     = (const float*)d_in[9];
    const float* W_hh     = (const float*)d_in[10];
    const float* b_ih     = (const float*)d_in[11];
    const float* b_hh     = (const float*)d_in[12];
    const float* ln_m_g   = (const float*)d_in[13];
    const float* ln_m_b   = (const float*)d_in[14];
    const float* W1       = (const float*)d_in[15];
    const float* b1       = (const float*)d_in[16];
    const float* W2       = (const float*)d_in[17];
    const float* b2       = (const float*)d_in[18];

    _Float16* kbuf = (_Float16*)d_ws;
    _Float16* vbuf = kbuf + (size_t)BN * NP * DD;   // 2 x 32 MB f16 in workspace

    dim3 g1(NP / 128, BN);
    kv_proj_kernel<<<g1, 256, 0, stream>>>(x, ln_in_g, ln_in_b, Wk, Wv, kbuf, vbuf);

    slot_iter_kernel<<<BN, 256, 0, stream>>>(kbuf, vbuf, slots_mu,
                                             ln_s_g, ln_s_b, Wq,
                                             W_ih, W_hh, b_ih, b_hh,
                                             ln_m_g, ln_m_b, W1, b1, W2, b2,
                                             (float*)d_out);
}
